// GRU_30064771072919
// MI455X (gfx1250) — matrix operand order repaired and checked by reading, not compiled, not run on hardware
//
#include <hip/hip_runtime.h>

// ---------------------------------------------------------------------------
// GRU forward for MI455X (gfx1250, wave32, WMMA) — persistent-kernel version.
//
// Roofline: ~309 GFLOP, ~200 MiB HBM traffic (9 us @ 23.3 TB/s): the
// recurrence is latency-bound, not bandwidth-bound. bf16 weights (9 MiB,
// B-fragment pre-packed) stay L2-resident across all 512 steps. The whole
// recurrence runs in ONE kernel (16 blocks = NH/64, 512 thr = 16 waves);
// steps are separated by a device-wide atomic barrier instead of 1024 kernel
// launches. The X@Wxh projection is computed before the mid-step barrier so
// only RH@Whh + tanh/combine sit on the serialized critical path.
// ---------------------------------------------------------------------------

#define T_STEPS 512
#define BATCH   64
#define NI_DIM  512
#define NH_DIM  1024
#define NBLOCKS 16           // NH/64 column-tile blocks (all co-resident)

typedef __attribute__((ext_vector_type(16))) __bf16        v16bf;
typedef __attribute__((ext_vector_type(8)))  float         v8f;
typedef __attribute__((ext_vector_type(4)))  unsigned int  u32x4;

union Frag16 { v16bf v; unsigned int u[8]; u32x4 q[2]; };
union Acc8   { v8f v; float f[8]; };

__device__ __forceinline__ unsigned short f32_to_bf16(float x) {
  unsigned int u = __float_as_uint(x);
  u += 0x7FFFu + ((u >> 16) & 1u);           // round-to-nearest-even
  return (unsigned short)(u >> 16);
}

__device__ __forceinline__ float sigmoid_fast(float x) {
  return 1.0f / (1.0f + __expf(-x));
}

// A fragment (16x32 bf16): lane l -> row l%16, K = 16 contiguous elements
// relative to kb + (l/16)*16.
__device__ __forceinline__ v16bf load_a_frag(const unsigned short* rowp) {
  Frag16 f;
  const unsigned int* p = (const unsigned int*)rowp;
  f.u[0] = p[0]; f.u[1] = p[1]; f.u[2] = p[2]; f.u[3] = p[3];
  f.u[4] = p[4]; f.u[5] = p[5]; f.u[6] = p[6]; f.u[7] = p[7];
  return f.v;
}

// B fragment: pre-packed, lane-contiguous 16 bf16 (32 bytes).
__device__ __forceinline__ v16bf load_b_frag(const unsigned short* lanep) {
  Frag16 f;
  const u32x4* q = (const u32x4*)lanep;
  f.q[0] = q[0]; f.q[1] = q[1];
  return f.v;
}

// Device-wide barrier: monotonic counter, release/acquire fences around it.
__device__ __forceinline__ void grid_sync(unsigned int* bar, unsigned int& target) {
  __threadfence();                 // release: my stores visible at agent scope
  __syncthreads();
  if (threadIdx.x == 0) {
    __hip_atomic_fetch_add(bar, 1u, __ATOMIC_RELEASE, __HIP_MEMORY_SCOPE_AGENT);
    while (__hip_atomic_load(bar, __ATOMIC_ACQUIRE, __HIP_MEMORY_SCOPE_AGENT) < target)
      __builtin_amdgcn_s_sleep(2);
    target += NBLOCKS;
  }
  __syncthreads();
  __threadfence();                 // acquire: drop stale lines before reading
}

// ---------------------------------------------------------------------------
// Pack f32 weight [K x NH] into WMMA B-fragment tiles (32x16 per tile).
// tile = ktile*(NH/16)+ntile; lane l holds (K = ktile*32 + (l/16)*16 + e,
// N = ntile*16 + l%16), e = 0..15, contiguous 32 bytes per lane.
// ---------------------------------------------------------------------------
__global__ void pack_weight_bf16(const float* __restrict__ W,
                                 unsigned short* __restrict__ dst, int K) {
  int gid  = blockIdx.x * blockDim.x + threadIdx.x;
  int tile = gid >> 5;
  int lane = gid & 31;
  const int ntiles = NH_DIM / 16;                 // 64
  int ktile = tile / ntiles;
  int ntile = tile - ktile * ntiles;
  if (ktile >= K / 32) return;
  int n     = ntile * 16 + (lane & 15);
  int kbase = ktile * 32 + ((lane >> 4) << 4);
  unsigned short* o = dst + ((size_t)tile << 9) + (lane << 4);
#pragma unroll
  for (int e = 0; e < 16; ++e)
    o[e] = f32_to_bf16(W[(size_t)(kbase + e) * NH_DIM + n]);
}

__global__ void convert_f32_bf16(const float* __restrict__ src,
                                 unsigned short* __restrict__ dst, int n) {
  int i = blockIdx.x * blockDim.x + threadIdx.x;
  if (i < n) dst[i] = f32_to_bf16(src[i]);
}

__global__ void zero_state(float* __restrict__ Hf,
                           unsigned short* __restrict__ Hbf,
                           unsigned int* __restrict__ bar) {
  int i = blockIdx.x * blockDim.x + threadIdx.x;
  Hf[i]  = 0.0f;
  Hbf[i] = 0;
  if (i == 0) *bar = 0u;
}

// ---------------------------------------------------------------------------
// Persistent GRU recurrence. Per step:
//   pre-barrier : az = X@Wxz + H@Whz ; ar = X@Wxr + H@Whr ; ah = X@Wxh
//                 Z = sig(az+bz) -> Zbuf ; RH = sig(ar+br)*H -> RHbf (bf16)
//   barrier     : RH fully materialized across all 16 blocks
//   post-barrier: ah += RH@Whh ; Ht = tanh(ah+bh) ; Hn = Z*H + (1-Z)*Ht
//                 -> outputs[t], Hf, Hbf
//   barrier     : H state ready for next step's K-reduction
// ---------------------------------------------------------------------------
__global__ void __launch_bounds__(512, 1)
gru_persistent(const unsigned short* __restrict__ Xbf,
               const unsigned short* __restrict__ Wxz,
               const unsigned short* __restrict__ Wxr,
               const unsigned short* __restrict__ Wxh,
               const unsigned short* __restrict__ Whz,
               const unsigned short* __restrict__ Whr,
               const unsigned short* __restrict__ Whh,
               const float* __restrict__ bz, const float* __restrict__ br,
               const float* __restrict__ bh,
               float* __restrict__ Hf, float* __restrict__ Zbuf,
               unsigned short* __restrict__ Hbf, unsigned short* __restrict__ RHbf,
               float* __restrict__ out, unsigned int* __restrict__ bar) {
  const int wave  = threadIdx.x >> 5;
  const int lane  = threadIdx.x & 31;
  const int mb    = (wave >> 2) << 4;                  // 0,16,32,48
  const int ntile = (blockIdx.x << 2) + (wave & 3);    // 0..63
  const int row   = lane & 15;
  const int khalf = (lane >> 4) << 4;
  const int bstep = (NH_DIM / 16) * 512;               // B-tile stride per kt
  const int n     = (ntile << 4) + (lane & 15);
  const size_t boff = ((size_t)ntile << 9) + (lane << 4);
  const float bzn = bz[n], brn = br[n], bhn = bh[n];

  unsigned int target = NBLOCKS;

  for (int t = 0; t < T_STEPS; ++t) {
    v8f az = {}, ar = {}, ah = {};

    // --- input projections (z, r, h), K = NI ---
    {
      const unsigned short* arow =
          Xbf + ((size_t)t * BATCH + mb + row) * NI_DIM + khalf;
      const unsigned short* pz = Wxz + boff;
      const unsigned short* pr = Wxr + boff;
      const unsigned short* ph = Wxh + boff;
#pragma unroll 4
      for (int kt = 0; kt < NI_DIM / 32; ++kt) {
        v16bf a  = load_a_frag(arow + kt * 32);
        v16bf fz = load_b_frag(pz);
        v16bf fr = load_b_frag(pr);
        v16bf fh = load_b_frag(ph);
        az = __builtin_amdgcn_wmma_f32_16x16x32_bf16(false, a, false, fz, (short)0, az, false, false);
        ar = __builtin_amdgcn_wmma_f32_16x16x32_bf16(false, a, false, fr, (short)0, ar, false, false);
        ah = __builtin_amdgcn_wmma_f32_16x16x32_bf16(false, a, false, fh, (short)0, ah, false, false);
        pz += bstep; pr += bstep; ph += bstep;
        __builtin_prefetch(pz, 0, 1);
        __builtin_prefetch(pr, 0, 1);
        __builtin_prefetch(ph, 0, 1);
      }
    }
    // --- recurrent projections (z, r), K = NH ---
    {
      const unsigned short* arow = Hbf + (size_t)(mb + row) * NH_DIM + khalf;
      const unsigned short* pz = Whz + boff;
      const unsigned short* pr = Whr + boff;
#pragma unroll 4
      for (int kt = 0; kt < NH_DIM / 32; ++kt) {
        v16bf a  = load_a_frag(arow + kt * 32);
        v16bf fz = load_b_frag(pz);
        v16bf fr = load_b_frag(pr);
        az = __builtin_amdgcn_wmma_f32_16x16x32_bf16(false, a, false, fz, (short)0, az, false, false);
        ar = __builtin_amdgcn_wmma_f32_16x16x32_bf16(false, a, false, fr, (short)0, ar, false, false);
        pz += bstep; pr += bstep;
        __builtin_prefetch(pz, 0, 1);
        __builtin_prefetch(pr, 0, 1);
      }
    }
    // --- epilogue A: Z, RH = sigmoid(r)*H ---
    {
      Acc8 cz, cr; cz.v = az; cr.v = ar;
#pragma unroll
      for (int r = 0; r < 8; ++r) {
        int m = mb + r + ((lane >> 4) << 3);
        size_t idx = (size_t)m * NH_DIM + n;
        float h  = Hf[idx];
        float z  = sigmoid_fast(cz.f[r] + bzn);
        float rg = sigmoid_fast(cr.f[r] + brn);
        Zbuf[idx] = z;
        RHbf[idx] = f32_to_bf16(rg * h);
      }
    }

    grid_sync(bar, target);            // RH complete across all blocks

    // --- candidate projection: ah += RH @ Whh, K = NH ---
    {
      const unsigned short* arow = RHbf + (size_t)(mb + row) * NH_DIM + khalf;
      const unsigned short* ph = Whh + boff;
#pragma unroll 4
      for (int kt = 0; kt < NH_DIM / 32; ++kt) {
        v16bf a = load_a_frag(arow + kt * 32);
        v16bf f = load_b_frag(ph);
        ah = __builtin_amdgcn_wmma_f32_16x16x32_bf16(false, a, false, f, (short)0, ah, false, false);
        ph += bstep;
        __builtin_prefetch(ph, 0, 1);
      }
    }
    // --- epilogue B: combine, write outputs + state ---
    {
      Acc8 c; c.v = ah;
#pragma unroll
      for (int r = 0; r < 8; ++r) {
        int m = mb + r + ((lane >> 4) << 3);
        size_t idx = (size_t)m * NH_DIM + n;
        float ht = tanhf(c.f[r] + bhn);
        float z  = Zbuf[idx];
        float h  = Hf[idx];
        float hn = z * h + (1.0f - z) * ht;
        out[(size_t)t * BATCH * NH_DIM + idx] = hn;
        Hf[idx]  = hn;
        Hbf[idx] = f32_to_bf16(hn);
      }
    }

    grid_sync(bar, target);            // H state complete for next step
  }
}

// ---------------------------------------------------------------------------
extern "C" void kernel_launch(void* const* d_in, const int* in_sizes, int n_in,
                              void* d_out, int out_size, void* d_ws, size_t ws_size,
                              hipStream_t stream) {
  (void)in_sizes; (void)n_in; (void)out_size; (void)ws_size;

  const float* inputs = (const float*)d_in[0];
  const float* W_xz = (const float*)d_in[1];
  const float* W_hz = (const float*)d_in[2];
  const float* b_z  = (const float*)d_in[3];
  const float* W_xr = (const float*)d_in[4];
  const float* W_hr = (const float*)d_in[5];
  const float* b_r  = (const float*)d_in[6];
  const float* W_xh = (const float*)d_in[7];
  const float* W_hh = (const float*)d_in[8];
  const float* b_h  = (const float*)d_in[9];
  float* out = (float*)d_out;

  // Workspace carve-out (256B aligned), total ~42 MiB.
  char* p = (char*)d_ws;
  auto carve = [&p](size_t bytes) -> void* {
    void* r = (void*)p;
    p += (bytes + 255) & ~(size_t)255;
    return r;
  };
  float*          Hf   = (float*)carve((size_t)BATCH * NH_DIM * 4);
  float*          Zbuf = (float*)carve((size_t)BATCH * NH_DIM * 4);
  unsigned short* Hbf  = (unsigned short*)carve((size_t)BATCH * NH_DIM * 2);
  unsigned short* RHbf = (unsigned short*)carve((size_t)BATCH * NH_DIM * 2);
  unsigned int*   bar  = (unsigned int*)carve(256);
  unsigned short* Xbf  = (unsigned short*)carve((size_t)T_STEPS * BATCH * NI_DIM * 2);
  unsigned short* Wxz  = (unsigned short*)carve((size_t)NI_DIM * NH_DIM * 2);
  unsigned short* Wxr  = (unsigned short*)carve((size_t)NI_DIM * NH_DIM * 2);
  unsigned short* Wxh  = (unsigned short*)carve((size_t)NI_DIM * NH_DIM * 2);
  unsigned short* Whz  = (unsigned short*)carve((size_t)NH_DIM * NH_DIM * 2);
  unsigned short* Whr  = (unsigned short*)carve((size_t)NH_DIM * NH_DIM * 2);
  unsigned short* Whh  = (unsigned short*)carve((size_t)NH_DIM * NH_DIM * 2);

  // Pack weights into B-fragment tiles (bf16).
  const int pk_x = (NI_DIM / 32) * (NH_DIM / 16) * 32 / 256;   // 128 blocks
  const int pk_h = (NH_DIM / 32) * (NH_DIM / 16) * 32 / 256;   // 256 blocks
  pack_weight_bf16<<<pk_x, 256, 0, stream>>>(W_xz, Wxz, NI_DIM);
  pack_weight_bf16<<<pk_x, 256, 0, stream>>>(W_xr, Wxr, NI_DIM);
  pack_weight_bf16<<<pk_x, 256, 0, stream>>>(W_xh, Wxh, NI_DIM);
  pack_weight_bf16<<<pk_h, 256, 0, stream>>>(W_hz, Whz, NH_DIM);
  pack_weight_bf16<<<pk_h, 256, 0, stream>>>(W_hr, Whr, NH_DIM);
  pack_weight_bf16<<<pk_h, 256, 0, stream>>>(W_hh, Whh, NH_DIM);

  // Convert inputs to bf16 once; zero the recurrent state + barrier counter.
  const int nx = T_STEPS * BATCH * NI_DIM;
  convert_f32_bf16<<<nx / 256, 256, 0, stream>>>(inputs, Xbf, nx);
  zero_state<<<(BATCH * NH_DIM) / 256, 256, 0, stream>>>(Hf, Hbf, bar);

  // Entire 512-step recurrence in a single persistent kernel.
  gru_persistent<<<NBLOCKS, 512, 0, stream>>>(Xbf, Wxz, Wxr, Wxh, Whz, Whr, Whh,
                                              b_z, b_r, b_h,
                                              Hf, Zbuf, Hbf, RHbf, out, bar);

  // H_last appended after outputs.
  hipMemcpyAsync(out + (size_t)T_STEPS * BATCH * NH_DIM, Hf,
                 (size_t)BATCH * NH_DIM * sizeof(float),
                 hipMemcpyDeviceToDevice, stream);
}
